// CANN_63479616635068
// MI455X (gfx1250) — matrix operand order repaired and checked by reading, compiled, not run on hardware
//
#include <hip/hip_runtime.h>
#include <hip/hip_bf16.h>
#include <stdint.h>

// ---------------------------------------------------------------------------
// Neural-CA update step for MI455X (gfx1250, wave32, WMMA).
//   perceived = depthwise [identity|sobel_x|sobel_y]  (K = 48, padded to 64)
//   h  = relu(w1 @ perceived + b1)   : V_WMMA_F32_16X16X32_F16, A=w1 tiles
//   dx = w2 @ h                      : V_WMMA_F32_16X16X32_F16, A=w2 tiles
//   Xn = X + dx * (rand < 0.5)
//   out = Xn * (maxpool3(X.a)>0.1 & maxpool3(Xn.a)>0.1)
// ---------------------------------------------------------------------------

typedef __attribute__((ext_vector_type(16))) _Float16 v16h;
typedef __attribute__((ext_vector_type(8)))  float    v8f;
typedef __attribute__((ext_vector_type(8)))  int      v8i;
typedef __attribute__((ext_vector_type(2)))  _Float16 h2t;

#define CCH 16
#define HIDN 128
#define NB  32
#define HH  256
#define WW  256
#define HW  (HH*WW)
#define CHW (CCH*HW)
#define KIN 48                  // real perceived features (padded to 64)

// workspace layout (bytes)
#define WS_W1     0             // 16 frags x 1024 B (A-layout f16, w1)
#define WS_W2     16384         // 4 frags x 1024 B (A-layout f16, w2)
#define WS_WBYTES 20480
#define WS_MASK   32768         // NB*HW bytes live-mask

// ---------------------------------------------------------------------------
// Pack w1/w2 (f32) into WMMA 16x16x32 f16 A-fragment layout.
// k order used here (contiguous, matching the B fragments): lanes 0-15 hold K {0..15}, lanes 16-31
// hold K {16..31}; element e of the per-lane v16h maps to
// Koff = e + 16*(lane>>4).
// ---------------------------------------------------------------------------
__global__ void __launch_bounds__(256)
nca_prep_weights(const float* __restrict__ w1, const float* __restrict__ w2,
                 _Float16* __restrict__ wfrag) {
  int t = blockIdx.x * blockDim.x + threadIdx.x;
  if (t >= 20 * 32) return;
  int frag  = t >> 5;
  int lane  = t & 31;
  int m     = lane & 15;
  int khalf = lane >> 4;
  _Float16* dst = wfrag + frag * 512 + lane * 16;
#pragma unroll
  for (int e = 0; e < 16; ++e) {
    int koff = e + 16 * khalf;
    float v;
    if (frag < 16) {                      // w1 : frag = hidtile*2 + kstep
      int tt = frag >> 1, s = frag & 1;
      int row = tt * 16 + m;              // hidden row
      int k   = s * 32 + koff;            // perceived feature (pad >=48 -> 0)
      v = (k < KIN) ? w1[row * KIN + k] : 0.f;
    } else {                              // w2 : frag-16 = kstep over hidden
      int s = frag - 16;
      int k = s * 32 + koff;
      v = w2[m * HIDN + k];
    }
    dst[e] = (_Float16)v;
  }
}

// pack two f32 into packed-f16 dword (v_cvt_pk_f16_f32)
__device__ __forceinline__ int pkh2(float a, float b) {
  h2t t = { (_Float16)a, (_Float16)b };
  return __builtin_bit_cast(int, t);
}
// value from lane^16 (swap wave halves): ds_swizzle xor=0x10, and=0x1f
__device__ __forceinline__ int xswap16(int v) {
  return __builtin_amdgcn_ds_swizzle(v, 0x401f);
}

// ---------------------------------------------------------------------------
// Kernel A: perception + MLP + stochastic residual. One wave = 16 pixels.
// ---------------------------------------------------------------------------
__global__ void __launch_bounds__(256)
nca_update(const float* __restrict__ X, const float* __restrict__ b1v,
           const float* __restrict__ rand_u,
           const _Float16* __restrict__ wfrag, float* __restrict__ out) {
  __shared__ __align__(32) _Float16 sW[WS_WBYTES / 2];
  __shared__ __align__(16) float sB[HIDN];

  {   // stage weight fragments (20 KB) + bias (512 B) into LDS
    const uint4* src = (const uint4*)wfrag;
    uint4*       dst = (uint4*)sW;
    for (int i = threadIdx.x; i < WS_WBYTES / 16; i += blockDim.x) dst[i] = src[i];
    if (threadIdx.x < HIDN) sB[threadIdx.x] = b1v[threadIdx.x];
  }
  __syncthreads();

  const int  lane = threadIdx.x & 31;
  const int  wave = threadIdx.x >> 5;
  const bool hi   = lane >= 16;
  const int  lx   = lane & 15;
  const int  bsel = hi ? 8 : 0;

  const int groupsTotal = NB * HW / 16;
  const int nWaves      = gridDim.x * (blockDim.x >> 5);

  for (int g = blockIdx.x * (blockDim.x >> 5) + wave; g < groupsTotal; g += nWaves) {
    int p0 = g << 4;
    int b  = p0 / HW;
    int rr = p0 - b * HW;
    int y  = rr >> 8;
    int x  = (rr & 255) + lx;
    const float* Xb = X + b * CHW;

    // ---- perception into B fragments (N=pixel on lane, K on elements) ----
    // B1[0]: lanes<16 K0..15 = identity c, lanes>=16 K16..31 = sobel_x c
    // B1[1]: lanes<16 K32..47 = sobel_y c, lanes>=16 K48..63 = zero pad
    v8i b1i0, b1i1;
    {
      int  ym = y - 1, yp = y + 1, xm = x - 1, xp = x + 1;
      bool okm = ym >= 0, okp = yp < HH, okl = xm >= 0, okr = xp < WW;
#pragma unroll
      for (int j = 0; j < 8; ++j) {        // two channels per step
        float v0[2], v1[2];
#pragma unroll
        for (int u = 0; u < 2; ++u) {
          int c = 2 * j + u;
          const float* base = Xb + c * HW;
          const float* rm = base + ym * WW;
          const float* rc = base + y  * WW;
          const float* rp = base + yp * WW;
          float n_mm = (okm && okl) ? rm[xm] : 0.f;
          float n_m0 = okm          ? rm[x]  : 0.f;
          float n_mp = (okm && okr) ? rm[xp] : 0.f;
          float n_0m = okl          ? rc[xm] : 0.f;
          float n_00 =                rc[x];
          float n_0p = okr          ? rc[xp] : 0.f;
          float n_pm = (okp && okl) ? rp[xm] : 0.f;
          float n_p0 = okp          ? rp[x]  : 0.f;
          float n_pp = (okp && okr) ? rp[xp] : 0.f;
          float sx = (n_mp - n_mm + 2.f * (n_0p - n_0m) + n_pp - n_pm) * 0.125f;
          float sy = (n_pm - n_mm + 2.f * (n_p0 - n_m0) + n_pp - n_mp) * 0.125f;
          v0[u] = hi ? sx : n_00;
          v1[u] = hi ? 0.f : sy;
        }
        b1i0[j] = pkh2(v0[0], v0[1]);
        b1i1[j] = pkh2(v1[0], v1[1]);
        // cap load hoisting: keep register liveness bounded
        __builtin_amdgcn_sched_barrier(0);
      }
    }
    v16h b1_0 = __builtin_bit_cast(v16h, b1i0);
    v16h b1_1 = __builtin_bit_cast(v16h, b1i1);

    // ---- layer 1 (8 hidden tiles x 2 ksteps) fused with layer-2 B build ----
    v8f acc2 = {};
#pragma unroll
    for (int s = 0; s < 4; ++s) {
      int q[2][4], p[2][4];   // packed-f16 hidden pairs: own / partner half
#pragma unroll
      for (int u = 0; u < 2; ++u) {
        int t = 2 * s + u;
        v8f  acc = {};
        v16h a;
        a   = *(const v16h*)&sW[(t * 2 + 0) * 512 + lane * 16];
        acc = __builtin_amdgcn_wmma_f32_16x16x32_f16(false, a, false, b1_0,
                                                     (short)0, acc, false, false);
        a   = *(const v16h*)&sW[(t * 2 + 1) * 512 + lane * 16];
        acc = __builtin_amdgcn_wmma_f32_16x16x32_f16(false, a, false, b1_1,
                                                     (short)0, acc, false, false);
        const float4* bp = (const float4*)&sB[t * 16 + bsel];
        float4 ba = bp[0], bb = bp[1];
        float h0 = fmaxf(acc[0] + ba.x, 0.f), h1 = fmaxf(acc[1] + ba.y, 0.f);
        float hv2 = fmaxf(acc[2] + ba.z, 0.f), h3 = fmaxf(acc[3] + ba.w, 0.f);
        float h4 = fmaxf(acc[4] + bb.x, 0.f), h5 = fmaxf(acc[5] + bb.y, 0.f);
        float h6 = fmaxf(acc[6] + bb.z, 0.f), h7 = fmaxf(acc[7] + bb.w, 0.f);
        q[u][0] = pkh2(h0, h1);
        q[u][1] = pkh2(hv2, h3);
        q[u][2] = pkh2(h4, h5);
        q[u][3] = pkh2(h6, h7);
#pragma unroll
        for (int jj = 0; jj < 4; ++jj) p[u][jj] = xswap16(q[u][jj]);
      }
      // D1 (hid on vgpr+lane-half) -> B2 layout (lanes<16: K..+15, >=16: +16..31)
      v8i b2i;
#pragma unroll
      for (int jj = 0; jj < 4; ++jj) {
        b2i[jj]     = hi ? p[1][jj] : q[0][jj];
        b2i[jj + 4] = hi ? q[1][jj] : p[0][jj];
      }
      v16h b2 = __builtin_bit_cast(v16h, b2i);
      v16h a2 = *(const v16h*)&sW[WS_W2 / 2 + s * 512 + lane * 16];
      acc2 = __builtin_amdgcn_wmma_f32_16x16x32_f16(false, a2, false, b2,
                                                    (short)0, acc2, false, false);
    }

    // ---- stochastic residual: lane half h owns channels 8h..8h+7 ----
    float ru  = rand_u[b * HW + y * WW + x];
    float msk = (ru < 0.5f) ? 1.f : 0.f;
    int   ob  = b * CHW + bsel * HW + y * WW + x;
#pragma unroll
    for (int r = 0; r < 8; ++r)
      out[ob + r * HW] = X[ob + r * HW] + acc2[r] * msk;
  }
}

// ---------------------------------------------------------------------------
// Kernel B: live mask = (maxpool3(X.alpha)>0.1) & (maxpool3(Xn.alpha)>0.1)
// ---------------------------------------------------------------------------
__global__ void __launch_bounds__(256)
nca_livemask(const float* __restrict__ X, const float* __restrict__ Xn,
             unsigned char* __restrict__ mask) {
  int p = blockIdx.x * blockDim.x + threadIdx.x;
  if (p >= NB * HW) return;
  int b = p / HW;
  int rr = p - b * HW;
  int y = rr >> 8, x = rr & 255;
  const float* a0 = X  + b * CHW + 3 * HW;
  const float* a1 = Xn + b * CHW + 3 * HW;
  float m0 = -1e30f, m1 = -1e30f;
#pragma unroll
  for (int dy = -1; dy <= 1; ++dy) {
    int yy = y + dy;
    if (yy < 0 || yy >= HH) continue;
#pragma unroll
    for (int dx = -1; dx <= 1; ++dx) {
      int xx = x + dx;
      if (xx < 0 || xx >= WW) continue;
      int o = yy * WW + xx;
      m0 = fmaxf(m0, a0[o]);
      m1 = fmaxf(m1, a1[o]);
    }
  }
  mask[p] = (m0 > 0.1f && m1 > 0.1f) ? 1 : 0;
}

// ---------------------------------------------------------------------------
// Kernel C: out *= live (float4 vectorized, 4 pixels -> uchar4 mask)
// ---------------------------------------------------------------------------
__global__ void __launch_bounds__(256)
nca_applymask(float* __restrict__ Xn, const unsigned char* __restrict__ mask) {
  int tid = blockIdx.x * blockDim.x + threadIdx.x;
  int i = tid * 4;
  if (i >= NB * CHW) return;
  int b = i / CHW;
  int o = i % HW;                    // pixel offset within image
  int pix = b * HW + o;
  float4 v = *(float4*)(Xn + i);
  uchar4 m = *(const uchar4*)(mask + pix);
  v.x *= (float)m.x;
  v.y *= (float)m.y;
  v.z *= (float)m.z;
  v.w *= (float)m.w;
  *(float4*)(Xn + i) = v;
}

// ---------------------------------------------------------------------------
extern "C" void kernel_launch(void* const* d_in, const int* in_sizes, int n_in,
                              void* d_out, int out_size, void* d_ws, size_t ws_size,
                              hipStream_t stream) {
  const float* X      = (const float*)d_in[0];
  const float* w1     = (const float*)d_in[1];
  const float* b1     = (const float*)d_in[2];
  const float* w2     = (const float*)d_in[3];
  const float* rand_u = (const float*)d_in[4];
  float*       out    = (float*)d_out;

  _Float16*      wfrag = (_Float16*)d_ws;
  unsigned char* mask  = (unsigned char*)d_ws + WS_MASK;

  // 1) pack weights into WMMA A-fragment layout
  nca_prep_weights<<<3, 256, 0, stream>>>(w1, w2, wfrag);
  // 2) perception + MLP + stochastic residual (writes all of d_out)
  nca_update<<<2048, 256, 0, stream>>>(X, b1, rand_u, wfrag, out);
  // 3) live mask from pre/post alpha neighborhoods
  nca_livemask<<<(NB * HW) / 256, 256, 0, stream>>>(X, out, mask);
  // 4) in-place scale by live mask
  nca_applymask<<<(NB * CHW) / (256 * 4), 256, 0, stream>>>(out, mask);
}
